// ScaledLinear_40896678592537
// MI455X (gfx1250) — 8-bit matrix operand order repaired and checked by reading, compiled, not run on hardware
//
#include <hip/hip_runtime.h>


typedef __attribute__((ext_vector_type(16))) int   v16i;
typedef __attribute__((ext_vector_type(8)))  float v8f;

#define FP8_MAX 448.0f

#if defined(__AMDGCN__) && \
    __has_builtin(__builtin_amdgcn_global_load_async_to_lds_b128) && \
    __has_builtin(__builtin_amdgcn_s_wait_asynccnt)
#define USE_ASYNC_LDS 1
#else
#define USE_ASYNC_LDS 0
#endif

__device__ __forceinline__ unsigned int f32_to_e4m3(float x) {
    unsigned int xb   = __float_as_uint(x);
    unsigned int sign = (xb >> 24) & 0x80u;
    float ax = fminf(fabsf(x), FP8_MAX);
    unsigned int out;
    if (ax < 0.015625f) {
        out = (unsigned int)rintf(ax * 512.0f);
    } else {
        unsigned int ab    = __float_as_uint(ax);
        unsigned int m     = ab & 0x7FFFFFu;
        unsigned int r     = m + 0x7FFFFu + ((m >> 20) & 1u);
        unsigned int carry = r >> 23;
        unsigned int mant  = (r >> 20) & 7u;
        unsigned int e     = (ab >> 23) - 120u + carry;
        out = (e << 3) | mant;
    }
    return out | sign;
}

__global__ void init_scales_kernel(unsigned int* scales) {
    if (threadIdx.x == 0) { scales[0] = 0u; scales[1] = 0u; }
}

__global__ void amax_kernel(const float4* __restrict__ in, long long n4,
                            unsigned int* __restrict__ out) {
    float m = 0.0f;
    long long stride = (long long)gridDim.x * blockDim.x;
    for (long long i = (long long)blockIdx.x * blockDim.x + threadIdx.x;
         i < n4; i += stride) {
        float4 v = in[i];
        m = fmaxf(m, fmaxf(fmaxf(fabsf(v.x), fabsf(v.y)),
                           fmaxf(fabsf(v.z), fabsf(v.w))));
    }
    #pragma unroll
    for (int off = 16; off > 0; off >>= 1)
        m = fmaxf(m, __shfl_xor(m, off, 32));
    if ((threadIdx.x & 31) == 0)
        atomicMax(out, __float_as_uint(m));
}

__global__ void quant_kernel(const float4* __restrict__ in,
                             unsigned int* __restrict__ q, long long n4,
                             const unsigned int* __restrict__ amax_bits) {
    float amax = __uint_as_float(*amax_bits);
    float inv  = fmaxf(amax * (1.0f / FP8_MAX), 1e-12f);
    float s    = 1.0f / inv;
    long long stride = (long long)gridDim.x * blockDim.x;
    for (long long i = (long long)blockIdx.x * blockDim.x + threadIdx.x;
         i < n4; i += stride) {
        float4 v = in[i];
        unsigned int p =  f32_to_e4m3(v.x * s)
                       | (f32_to_e4m3(v.y * s) << 8)
                       | (f32_to_e4m3(v.z * s) << 16)
                       | (f32_to_e4m3(v.w * s) << 24);
        q[i] = p;
    }
}

#define WM 2
#define WN 4

#if USE_ASYNC_LDS
typedef int v4i_raw __attribute__((vector_size(16)));
typedef __attribute__((address_space(1))) v4i_raw* gptr_b128;
typedef __attribute__((address_space(3))) v4i_raw* lptr_b128;

__device__ __forceinline__ void stage_b_async(const unsigned int* qb,
                                              unsigned int* ldsDst,
                                              int blockN, int K4, int kb,
                                              int tid) {
    const int row  = tid >> 1;
    const int half = (tid & 1) * 16;
    const unsigned int* g = qb + (long long)(blockN + row) * K4 + kb + half;
    unsigned int* l = ldsDst + row * 32 + half;
    #pragma unroll
    for (int j = 0; j < 4; ++j) {
        __builtin_amdgcn_global_load_async_to_lds_b128(
            (gptr_b128)(g + j * 4),
            (lptr_b128)(l + j * 4),
            0, 0);
    }
}
#endif

__global__ __launch_bounds__(256)
void gemm_fp8_kernel(const unsigned int* __restrict__ qa,
                     const unsigned int* __restrict__ qb,
                     const float* __restrict__ bias,
                     const unsigned int* __restrict__ scales,
                     float* __restrict__ out,
                     int M, int N, int K) {
    const int K4   = K >> 2;
    const int tid  = threadIdx.x;
    const int wave = tid >> 5;
    const int lane = tid & 31;
    const int h    = lane >> 4;
    const int l16  = lane & 15;
    const int wm   = wave >> 1;
    const int wn   = wave & 1;

    const int blockM = blockIdx.x * 128;
    const int blockN = blockIdx.y * 128;

    const int rowA = blockM + wm * (WM * 16) + l16;

    const unsigned int* aRow[WM];
    #pragma unroll
    for (int tm = 0; tm < WM; ++tm)
        aRow[tm] = qa + (long long)(rowA + tm * 16) * K4;

    v8f acc[WM][WN] = {};

#if USE_ASYNC_LDS
    __shared__ unsigned int ldsB[2][128 * 32];

    const int steps = K4 >> 5;
    stage_b_async(qb, &ldsB[0][0], blockN, K4, 0, tid);

    for (int i = 0; i < steps; ++i) {
        const int kb = i << 5;
        if (i + 1 < steps) {
            stage_b_async(qb, &ldsB[(i + 1) & 1][0], blockN, K4, kb + 32, tid);
            __builtin_amdgcn_s_wait_asynccnt(4);
        } else {
            __builtin_amdgcn_s_wait_asynccnt(0);
        }
        __syncthreads();

        const unsigned int* lb = &ldsB[i & 1][0];

        v16i af[WM], bf[WN];
        #pragma unroll
        for (int tm = 0; tm < WM; ++tm) {
            #pragma unroll
            for (int p = 0; p < 8; ++p) {
                int off = kb + (p >> 1) * 8 + h * 4 + (p & 1) * 2;
                uint2 d = *(const uint2*)(aRow[tm] + off);
                af[tm][2 * p]     = (int)d.x;
                af[tm][2 * p + 1] = (int)d.y;
            }
        }
        #pragma unroll
        for (int tn = 0; tn < WN; ++tn) {
            const unsigned int* lrow = lb + (wn * 64 + tn * 16 + l16) * 32;
            #pragma unroll
            for (int g = 0; g < 4; ++g) {
                uint4 d = *(const uint4*)(lrow + g * 8 + h * 4);
                bf[tn][4 * g]     = (int)d.x;
                bf[tn][4 * g + 1] = (int)d.y;
                bf[tn][4 * g + 2] = (int)d.z;
                bf[tn][4 * g + 3] = (int)d.w;
            }
        }
        #pragma unroll
        for (int tm = 0; tm < WM; ++tm)
            #pragma unroll
            for (int tn = 0; tn < WN; ++tn)
                acc[tm][tn] = __builtin_amdgcn_wmma_f32_16x16x128_fp8_fp8(
                    af[tm], bf[tn], (short)0, acc[tm][tn], false, false);

        __syncthreads();
    }
#else
    const int colB = blockN + wn * (WN * 16) + l16;
    const unsigned int* bRow[WN];
    #pragma unroll
    for (int tn = 0; tn < WN; ++tn)
        bRow[tn] = qb + (long long)(colB + tn * 16) * K4;

    for (int kb = 0; kb < K4; kb += 32) {
        v16i af[WM], bf[WN];
        #pragma unroll
        for (int tm = 0; tm < WM; ++tm) {
            #pragma unroll
            for (int p = 0; p < 8; ++p) {
                int off = kb + (p >> 1) * 8 + h * 4 + (p & 1) * 2;
                uint2 d = *(const uint2*)(aRow[tm] + off);
                af[tm][2 * p]     = (int)d.x;
                af[tm][2 * p + 1] = (int)d.y;
            }
        }
        #pragma unroll
        for (int tn = 0; tn < WN; ++tn) {
            #pragma unroll
            for (int g = 0; g < 4; ++g) {
                int off = kb + g * 8 + h * 4;
                uint4 d = *(const uint4*)(bRow[tn] + off);
                bf[tn][4 * g]     = (int)d.x;
                bf[tn][4 * g + 1] = (int)d.y;
                bf[tn][4 * g + 2] = (int)d.z;
                bf[tn][4 * g + 3] = (int)d.w;
            }
        }
        #pragma unroll
        for (int tm = 0; tm < WM; ++tm)
            #pragma unroll
            for (int tn = 0; tn < WN; ++tn)
                acc[tm][tn] = __builtin_amdgcn_wmma_f32_16x16x128_fp8_fp8(
                    af[tm], bf[tn], (short)0, acc[tm][tn], false, false);
    }
#endif

    float invx = fmaxf(__uint_as_float(scales[0]) * (1.0f / FP8_MAX), 1e-12f);
    float invw = fmaxf(__uint_as_float(scales[1]) * (1.0f / FP8_MAX), 1e-12f);
    float sc = invx * invw;

    const int rBase = blockM + wm * (WM * 16);
    const int cBase = blockN + wn * (WN * 16);
    #pragma unroll
    for (int tm = 0; tm < WM; ++tm) {
        #pragma unroll
        for (int tn = 0; tn < WN; ++tn) {
            int col  = cBase + tn * 16 + l16;
            float bv = bias[col];
            #pragma unroll
            for (int r = 0; r < 8; ++r) {
                int row = rBase + tm * 16 + r + h * 8;
                out[(long long)row * N + col] = acc[tm][tn][r] * sc + bv;
            }
        }
    }
}

extern "C" void kernel_launch(void* const* d_in, const int* in_sizes, int n_in,
                              void* d_out, int out_size, void* d_ws, size_t ws_size,
                              hipStream_t stream) {
    const float* x    = (const float*)d_in[0];
    const float* w    = (const float*)d_in[1];
    const float* bias = (const float*)d_in[2];

    const int N = in_sizes[2];
    const int K = in_sizes[1] / N;
    const int M = in_sizes[0] / K;

    unsigned int* scales = (unsigned int*)d_ws;
    unsigned int* qx = (unsigned int*)((char*)d_ws + 256);
    unsigned int* qw = qx + (size_t)M * K / 4;

    init_scales_kernel<<<1, 32, 0, stream>>>(scales);

    long long nx4 = (long long)M * K / 4;
    long long nw4 = (long long)N * K / 4;
    amax_kernel<<<1024, 256, 0, stream>>>((const float4*)x, nx4, &scales[0]);
    amax_kernel<<<512,  256, 0, stream>>>((const float4*)w, nw4, &scales[1]);

    quant_kernel<<<2048, 256, 0, stream>>>((const float4*)x, qx, nx4, &scales[0]);
    quant_kernel<<<1024, 256, 0, stream>>>((const float4*)w, qw, nw4, &scales[1]);

    dim3 grid((M + 127) / 128, (N + 127) / 128);
    gemm_fp8_kernel<<<grid, 256, 0, stream>>>(qx, qw, bias, scales,
                                              (float*)d_out, M, N, K);
}
